// RotaryAttention_16801912062198
// MI455X (gfx1250) — compile-verified, matrix operand order checked by reading, not run on hardware
//
#include <hip/hip_runtime.h>

// ---------------- problem constants ----------------
#define B_   4
#define S_   1024
#define D_   2048
#define H_   16
#define G_   2
#define HD_  128
#define MTOK (B_ * S_)   // 4096 rows

typedef __bf16 bf16_t;
typedef bf16_t v16bf __attribute__((ext_vector_type(16)));
typedef float  v8f   __attribute__((ext_vector_type(8)));
typedef int    i32x4 __attribute__((ext_vector_type(4)));

// gfx1250 async global->LDS copies (ASYNCcnt-tracked), guarded so the file
// compiles on toolchains where the builtin name differs (falls back to a
// register-staged software pipeline).
#if defined(__HIP_DEVICE_COMPILE__) && \
    __has_builtin(__builtin_amdgcn_global_load_async_to_lds_b128)
#define ASYNC_LDS 1
#else
#define ASYNC_LDS 0
#endif

// ---------------- helpers ----------------
__device__ __forceinline__ unsigned short f2bfu(float f) {
    unsigned int u = __float_as_uint(f);
    unsigned int r = u + 0x7fffu + ((u >> 16) & 1u);   // round-to-nearest-even
    return (unsigned short)(r >> 16);
}
__device__ __forceinline__ float bfu2f(unsigned short h) {
    return __uint_as_float(((unsigned int)h) << 16);
}

// 16-bit A-style WMMA fragment (16x32, M x K): lane L holds row (L&15),
// K elements [k8..k8+7] and [k8+16..k8+23] where k8 = 8*(L>=16).
// Same addressing serves B fragments when the matrix is stored N x K.
__device__ __forceinline__ v16bf load_frag(const unsigned short* rowPtr, int k8) {
    v16bf f;
    __builtin_memcpy(&f, rowPtr + k8, 16);
    __builtin_memcpy(reinterpret_cast<char*>(&f) + 16, rowPtr + k8 + 16, 16);
    return f;
}

__device__ __forceinline__ v8f wmma_bf16(v16bf a, v16bf b, v8f c) {
    // (neg_a, A, neg_b, B, c_mod, C, reuse_a, reuse_b)
    return __builtin_amdgcn_wmma_f32_16x16x32_bf16(false, a, false, b, (short)0, c,
                                                   false, false);
}

#if ASYNC_LDS
__device__ __forceinline__ void async_copy16(const unsigned short* g,
                                             unsigned short* l) {
    // signature (probe-revealed): (v4i addrspace(1)*, v4i addrspace(3)*, Ii, Ii)
    __builtin_amdgcn_global_load_async_to_lds_b128(
        (__attribute__((address_space(1))) i32x4*)g,
        (__attribute__((address_space(3))) i32x4*)l, 0, 0);
}
__device__ __forceinline__ void wait_async_le3() {
#if __has_builtin(__builtin_amdgcn_s_wait_asynccnt)
    __builtin_amdgcn_s_wait_asynccnt(3);
#else
    asm volatile("s_wait_asynccnt 0x3" ::: "memory");
#endif
}
__device__ __forceinline__ void wait_async_le0() {
#if __has_builtin(__builtin_amdgcn_s_wait_asynccnt)
    __builtin_amdgcn_s_wait_asynccnt(0);
#else
    asm volatile("s_wait_asynccnt 0x0" ::: "memory");
#endif
}
#endif

// ---------------- pre-pass: conversions ----------------
__global__ void cvt_f32_bf16(const float* __restrict__ in,
                             unsigned short* __restrict__ out, int n) {
    int i = blockIdx.x * 256 + threadIdx.x;
    if (i < n) out[i] = f2bfu(in[i]);
}

// in: K x N f32 (row-major)  ->  out: N x K bf16
__global__ void transpose_f32_bf16(const float* __restrict__ in,
                                   unsigned short* __restrict__ out,
                                   int K, int N) {
    long long i = (long long)blockIdx.x * 256 + threadIdx.x;
    long long total = (long long)K * N;
    if (i >= total) return;
    int n = (int)(i / K);
    int k = (int)(i % K);
    out[(long long)n * K + k] = f2bfu(in[(long long)k * N + n]);
}

// ---------------- bf16 WMMA GEMM: C[MxN] = A[MxK] * BT[NxK]^T + bias ----------------
// 256 threads = 8 waves (2 M-waves x 4 N-waves), block tile 64M x 128N,
// wave tile 32x32 (2x2 fragments), K step 32, double-buffered LDS.
// Staging: async global->LDS copies when available, else a register-staged
// software pipeline (loads for tile k+1 in flight while computing tile k).
template <int OUT_BF16>
__launch_bounds__(256)
__global__ void gemm_bf16_wmma(const unsigned short* __restrict__ A,
                               const unsigned short* __restrict__ BT,
                               const float* __restrict__ bias,
                               void* __restrict__ C,
                               int M, int N, int K) {
    __shared__ unsigned short As[2][64][40];     // 32 K + pad
    __shared__ unsigned short Bs[2][128][40];
    const int A_STR = 64 * 40;                   // shorts per buffer
    const int B_STR = 128 * 40;

    const int tid   = threadIdx.x;
    const int lane  = tid & 31;
    const int wid   = tid >> 5;
    const int wm    = wid & 1;
    const int wn    = wid >> 1;
    const int m0    = blockIdx.y * 64;
    const int n0    = blockIdx.x * 128;
    const int row16 = lane & 15;
    const int half  = lane >> 4;
    const int k8    = half * 8;

    v8f acc[2][2];
#pragma unroll
    for (int i = 0; i < 2; ++i)
#pragma unroll
        for (int j = 0; j < 2; ++j) acc[i][j] = {};

    // staging addresses: 1 A chunk + 2 B chunks of 16B per thread per tile
    const int arow = tid >> 2, ac8 = (tid & 3) * 8;
    const unsigned short* gA  = &A[(long long)(m0 + arow) * K + ac8];
    const unsigned short* gB0 = &BT[(long long)(n0 + arow) * K + ac8];
    const unsigned short* gB1 = &BT[(long long)(n0 + 64 + arow) * K + ac8];
    unsigned short* lA  = &As[0][arow][ac8];
    unsigned short* lB0 = &Bs[0][arow][ac8];
    unsigned short* lB1 = &Bs[0][64 + arow][ac8];

    const int niter = K >> 5;

#if ASYNC_LDS
    // prologue: stage tile 0 into buffer 0
    async_copy16(gA, lA);
    async_copy16(gB0, lB0);
    async_copy16(gB1, lB1);
#else
    float4 rA  = *(const float4*)gA;
    float4 rB0 = *(const float4*)gB0;
    float4 rB1 = *(const float4*)gB1;
    *(float4*)lA  = rA;
    *(float4*)lB0 = rB0;
    *(float4*)lB1 = rB1;
#endif

    for (int it = 0; it < niter; ++it) {
        const int cur = it & 1;
        const int nk  = (it + 1) << 5;
#if ASYNC_LDS
        if (it + 1 < niter) {
            const int nb = cur ^ 1;
            async_copy16(gA + nk,  lA  + nb * A_STR);
            async_copy16(gB0 + nk, lB0 + nb * B_STR);
            async_copy16(gB1 + nk, lB1 + nb * B_STR);
            wait_async_le3();          // drain tile `it`, keep tile `it+1` in flight
        } else {
            wait_async_le0();
        }
        __syncthreads();
#else
        float4 nA, nB0, nB1;
        if (it + 1 < niter) {
            nA  = *(const float4*)(gA + nk);
            nB0 = *(const float4*)(gB0 + nk);
            nB1 = *(const float4*)(gB1 + nk);
        }
        __syncthreads();
#endif

        v16bf aF[2], bF[2];
#pragma unroll
        for (int f = 0; f < 2; ++f)
            aF[f] = load_frag(&As[cur][wm * 32 + f * 16 + row16][0], k8);
#pragma unroll
        for (int f = 0; f < 2; ++f)
            bF[f] = load_frag(&Bs[cur][wn * 32 + f * 16 + row16][0], k8);
#pragma unroll
        for (int fm = 0; fm < 2; ++fm)
#pragma unroll
            for (int fn = 0; fn < 2; ++fn)
                acc[fm][fn] = wmma_bf16(aF[fm], bF[fn], acc[fm][fn]);

        __syncthreads();
#if !ASYNC_LDS
        if (it + 1 < niter) {
            const int nb = cur ^ 1;
            *(float4*)(lA + nb * A_STR)  = nA;
            *(float4*)(lB0 + nb * B_STR) = nB0;
            *(float4*)(lB1 + nb * B_STR) = nB1;
        }
#endif
    }

    // epilogue: C/D layout -> lane L holds col (L&15), rows r + 8*(L>=16)
#pragma unroll
    for (int fm = 0; fm < 2; ++fm)
#pragma unroll
        for (int fn = 0; fn < 2; ++fn) {
            int col = n0 + wn * 32 + fn * 16 + row16;
            float bv = bias[col];
#pragma unroll
            for (int r = 0; r < 8; ++r) {
                int row = m0 + wm * 32 + fm * 16 + r + 8 * half;
                float v = acc[fm][fn][r] + bv;
                if (OUT_BF16)
                    ((unsigned short*)C)[(long long)row * N + col] = f2bfu(v);
                else
                    ((float*)C)[(long long)row * N + col] = v;
            }
        }
}

// ---------------- RoPE (in place, bf16), optional q-scale ----------------
__global__ void rope_inplace(unsigned short* __restrict__ buf,
                             int heads, int seqLen, float scale, int total) {
    int i = blockIdx.x * 256 + threadIdx.x;
    if (i >= total) return;
    int c  = i & 63;            // pair index 0..63
    int th = i >> 6;            // t*heads + h
    int t  = th / heads;
    int pos = t % seqLen;
    // inv_freq[c] = 10000^(-2c/128)
    float invf = __expf((float)c * (-2.0f / 128.0f) * 9.210340371976184f);
    float ang  = (float)pos * invf;
    float cs = __cosf(ang), sn = __sinf(ang);
    long long base = (long long)th * 128;
    float x1 = bfu2f(buf[base + c]);
    float x2 = bfu2f(buf[base + c + 64]);
    buf[base + c]      = f2bfu((x1 * cs - x2 * sn) * scale);
    buf[base + c + 64] = f2bfu((x2 * cs + x1 * sn) * scale);
}

// ---------------- flash attention (GQA, no mask) ----------------
// block = (b, h, 64 q-rows); 128 threads = 4 waves, 16 q-rows per wave.
// Q fragments live in registers; K/V^T tiles (32 keys) staged in LDS;
// P re-laid out C->A through per-wave LDS scratch.
__launch_bounds__(128)
__global__ void flash_attn(const unsigned short* __restrict__ Q,   // MTOK x 2048
                           const unsigned short* __restrict__ Kb,  // MTOK x 256
                           const unsigned short* __restrict__ Vb,  // MTOK x 256
                           unsigned short* __restrict__ O) {       // MTOK x 2048
    __shared__ unsigned short Kt[32][136];      // keys x d (pad 8)
    __shared__ unsigned short Vt[128][40];      // d x keys (transposed V)
    __shared__ unsigned short Pt[4][16][40];    // per-wave P scratch

    const int tid  = threadIdx.x;
    const int lane = tid & 31;
    const int w    = tid >> 5;
    const int bid  = blockIdx.x;
    const int qb   = bid & 15;          // S/64 = 16 q blocks
    const int h    = (bid >> 4) & 15;
    const int b    = bid >> 8;
    const int g    = h >> 3;            // kv head = h / (H/G)

    const int row16 = lane & 15;
    const int half  = lane >> 4;
    const int k8    = half * 8;

    const int q0 = qb * 64 + w * 16;
    const unsigned short* qrow =
        Q + ((long long)(b * S_ + q0 + row16)) * 2048 + h * 128;

    v16bf qF[4];
#pragma unroll
    for (int kk = 0; kk < 4; ++kk) qF[kk] = load_frag(qrow + kk * 32, k8);

    v8f oAcc[8];
#pragma unroll
    for (int d = 0; d < 8; ++d) oAcc[d] = {};
    float mrow[8], lrow[8];
#pragma unroll
    for (int r = 0; r < 8; ++r) { mrow[r] = -1e30f; lrow[r] = 0.0f; }

    const float sscale = 0.08838834764831845f;  // hd^-0.5 (second factor)

    for (int kt = 0; kt < S_ / 32; ++kt) {
        const int n0 = kt * 32;
        __syncthreads();
        // stage K tile 32x128 row-major (contraction dim d contiguous)
#pragma unroll
        for (int i = 0; i < 4; ++i) {
            int c = tid + i * 128;              // chunk 0..511
            int r = c >> 4, c8 = (c & 15) * 8;
            *(float4*)&Kt[r][c8] =
                *(const float4*)&Kb[((long long)(b * S_ + n0 + r)) * 256 + g * 128 + c8];
        }
        // stage V transposed: Vt[d][key]
#pragma unroll
        for (int i = 0; i < 4; ++i) {
            int c = tid + i * 128;
            int r = c >> 4, c8 = (c & 15) * 8;
            unsigned short tmp[8];
            *(float4*)tmp =
                *(const float4*)&Vb[((long long)(b * S_ + n0 + r)) * 256 + g * 128 + c8];
#pragma unroll
            for (int j = 0; j < 8; ++j) Vt[c8 + j][r] = tmp[j];
        }
        // prefetch next tile into caches (global_prefetch_b8)
        if (kt + 1 < S_ / 32) {
            int r = tid >> 2, c8 = (tid & 3) * 32;
            __builtin_prefetch(
                &Kb[((long long)(b * S_ + n0 + 32 + r)) * 256 + g * 128 + c8], 0, 3);
            __builtin_prefetch(
                &Vb[((long long)(b * S_ + n0 + 32 + r)) * 256 + g * 128 + c8], 0, 3);
        }
        __syncthreads();

        // scores S = Q (16x128) x K^T (128x32): 2 N-frags x 4 K-steps
        v8f s[2];
#pragma unroll
        for (int nf = 0; nf < 2; ++nf) {
            v8f sa = {};
#pragma unroll
            for (int kk = 0; kk < 4; ++kk) {
                v16bf kF = load_frag(&Kt[nf * 16 + row16][0], kk * 32 + k8);
                sa = wmma_bf16(qF[kk], kF, sa);
            }
#pragma unroll
            for (int r = 0; r < 8; ++r) s[nf][r] = sa[r] * sscale;
        }

        // online softmax: rows live across 16-lane halves
        float rmax[8];
#pragma unroll
        for (int r = 0; r < 8; ++r) rmax[r] = fmaxf(s[0][r], s[1][r]);
#pragma unroll
        for (int d = 1; d < 16; d <<= 1)
#pragma unroll
            for (int r = 0; r < 8; ++r)
                rmax[r] = fmaxf(rmax[r], __shfl_xor(rmax[r], d, 32));

        float cf[8];
#pragma unroll
        for (int r = 0; r < 8; ++r) {
            float mn = fmaxf(mrow[r], rmax[r]);
            cf[r] = __expf(mrow[r] - mn);
            mrow[r] = mn;
        }
#pragma unroll
        for (int nf = 0; nf < 2; ++nf)
#pragma unroll
            for (int r = 0; r < 8; ++r) s[nf][r] = __expf(s[nf][r] - mrow[r]);

        float rsum[8];
#pragma unroll
        for (int r = 0; r < 8; ++r) rsum[r] = s[0][r] + s[1][r];
#pragma unroll
        for (int d = 1; d < 16; d <<= 1)
#pragma unroll
            for (int r = 0; r < 8; ++r) rsum[r] += __shfl_xor(rsum[r], d, 32);
#pragma unroll
        for (int r = 0; r < 8; ++r) lrow[r] = lrow[r] * cf[r] + rsum[r];

        // rescale O accumulators
#pragma unroll
        for (int d = 0; d < 8; ++d)
#pragma unroll
            for (int r = 0; r < 8; ++r) oAcc[d][r] *= cf[r];

        // P: C-layout -> A-layout via per-wave LDS
#pragma unroll
        for (int nf = 0; nf < 2; ++nf)
#pragma unroll
            for (int r = 0; r < 8; ++r)
                Pt[w][r + 8 * half][nf * 16 + row16] = f2bfu(s[nf][r]);
        __syncthreads();

        // O += P (16x32) x V (32x128)
        v16bf pF = load_frag(&Pt[w][row16][0], k8);
#pragma unroll
        for (int df = 0; df < 8; ++df) {
            v16bf vF = load_frag(&Vt[df * 16 + row16][0], k8);
            oAcc[df] = wmma_bf16(pF, vF, oAcc[df]);
        }
    }

    // finalize: O / l -> bf16
    unsigned short* orow = O + ((long long)(b * S_ + q0)) * 2048 + h * 128;
#pragma unroll
    for (int r = 0; r < 8; ++r) {
        float inv = 1.0f / lrow[r];
        int row = r + 8 * half;
#pragma unroll
        for (int df = 0; df < 8; ++df)
            orow[(long long)row * 2048 + df * 16 + row16] =
                f2bfu(oAcc[df][r] * inv);
    }
}

// ---------------- host launch ----------------
extern "C" void kernel_launch(void* const* d_in, const int* in_sizes, int n_in,
                              void* d_out, int out_size, void* d_ws, size_t ws_size,
                              hipStream_t stream) {
    (void)in_sizes; (void)n_in; (void)out_size; (void)ws_size;
    const float* x  = (const float*)d_in[0];
    const float* wq = (const float*)d_in[1];
    const float* bq = (const float*)d_in[2];
    const float* wk = (const float*)d_in[3];
    const float* bk = (const float*)d_in[4];
    const float* wv = (const float*)d_in[5];
    const float* bv = (const float*)d_in[6];
    const float* wo = (const float*)d_in[7];
    const float* bo = (const float*)d_in[8];

    char* ws = (char*)d_ws;
    size_t off = 0;
    auto alloc = [&](size_t bytes) -> void* {
        void* p = ws + off;
        off = (off + bytes + 255) & ~(size_t)255;
        return p;
    };
    unsigned short* xb  = (unsigned short*)alloc((size_t)MTOK * D_ * 2);      // 16 MB
    unsigned short* wqT = (unsigned short*)alloc((size_t)D_ * D_ * 2);        //  8 MB
    unsigned short* wkT = (unsigned short*)alloc((size_t)(G_ * HD_) * D_ * 2);
    unsigned short* wvT = (unsigned short*)alloc((size_t)(G_ * HD_) * D_ * 2);
    unsigned short* woT = (unsigned short*)alloc((size_t)D_ * D_ * 2);        //  8 MB
    unsigned short* Qb  = (unsigned short*)alloc((size_t)MTOK * D_ * 2);      // 16 MB
    unsigned short* Kb  = (unsigned short*)alloc((size_t)MTOK * (G_ * HD_) * 2);
    unsigned short* Vb  = (unsigned short*)alloc((size_t)MTOK * (G_ * HD_) * 2);
    unsigned short* Ob  = (unsigned short*)alloc((size_t)MTOK * D_ * 2);      // 16 MB

    const int KV = G_ * HD_;   // 256

    // 1) conversions
    {
        int n = MTOK * D_;
        cvt_f32_bf16<<<(n + 255) / 256, 256, 0, stream>>>(x, xb, n);
        long long t;
        t = (long long)D_ * D_;
        transpose_f32_bf16<<<(unsigned)((t + 255) / 256), 256, 0, stream>>>(wq, wqT, D_, D_);
        t = (long long)D_ * KV;
        transpose_f32_bf16<<<(unsigned)((t + 255) / 256), 256, 0, stream>>>(wk, wkT, D_, KV);
        transpose_f32_bf16<<<(unsigned)((t + 255) / 256), 256, 0, stream>>>(wv, wvT, D_, KV);
        t = (long long)D_ * D_;
        transpose_f32_bf16<<<(unsigned)((t + 255) / 256), 256, 0, stream>>>(wo, woT, D_, D_);
    }

    // 2) QKV projections (bf16 out)
    {
        dim3 gq(D_ / 128, MTOK / 64);
        gemm_bf16_wmma<1><<<gq, 256, 0, stream>>>(xb, wqT, bq, Qb, MTOK, D_, D_);
        dim3 gkv(KV / 128, MTOK / 64);
        gemm_bf16_wmma<1><<<gkv, 256, 0, stream>>>(xb, wkT, bk, Kb, MTOK, KV, D_);
        gemm_bf16_wmma<1><<<gkv, 256, 0, stream>>>(xb, wvT, bv, Vb, MTOK, KV, D_);
    }

    // 3) RoPE (q gets hd^-0.5 folded in)
    {
        int tq = MTOK * H_ * 64;
        rope_inplace<<<(tq + 255) / 256, 256, 0, stream>>>(
            Qb, H_, S_, 0.08838834764831845f, tq);
        int tk = MTOK * G_ * 64;
        rope_inplace<<<(tk + 255) / 256, 256, 0, stream>>>(Kb, G_, S_, 1.0f, tk);
    }

    // 4) flash attention
    {
        int nblk = B_ * H_ * (S_ / 64);   // 1024
        flash_attn<<<nblk, 128, 0, stream>>>(Qb, Kb, Vb, Ob);
    }

    // 5) output projection (f32 out -> d_out)
    {
        dim3 go(D_ / 128, MTOK / 64);
        gemm_bf16_wmma<0><<<go, 256, 0, stream>>>(Ob, woT, bo, (float*)d_out,
                                                  MTOK, D_, D_);
    }
}
